// MLPMoE_22917945492102
// MI455X (gfx1250) — matrix operand order repaired and checked by reading, not compiled, not run on hardware
//
#include <hip/hip_runtime.h>
#include <cstdint>

// ---------------------------------------------------------------------------
// MoE MLP for MI455X (gfx1250): LN -> router -> per-expert (D->H GELU H->H)
// Dense over E=4 experts (matches reference einsum), bf16 WMMA matmuls with
// GLOBAL_LOAD_ASYNC_TO_LDS_B128 double-buffered staging (ASYNCcnt).
// T=4608 tokens, D=1024, H=2048, E=4, K(top)=2.
// ---------------------------------------------------------------------------

#define T_TOK 4608
#define DD 1024
#define HH 2048
#define NE 4

typedef __attribute__((ext_vector_type(16))) __bf16 v16bf;
typedef __attribute__((ext_vector_type(8)))  float  v8f;

union FragBF {
    uint4  q[2];   // 2 x 16B = 32B
    v16bf  v;
};

static __device__ __forceinline__ uint16_t f2bf(float f) {
    // round-to-nearest-even fp32 -> bf16
    uint32_t u = __float_as_uint(f);
    u += 0x7FFFu + ((u >> 16) & 1u);
    return (uint16_t)(u >> 16);
}

static __device__ __forceinline__ float gelu_exact(float x) {
    return 0.5f * x * (1.0f + erff(x * 0.70710678118654752f));
}

// Flat pointer to __shared__ -> wave-relative LDS byte address (addr[31:0]).
static __device__ __forceinline__ uint32_t lds_addr32(const void* p) {
    return (uint32_t)(uintptr_t)p;
}

// gfx1250 async global->LDS copy, 16B per lane, tracked by ASYNCcnt.
static __device__ __forceinline__ void async_load_b128(uint32_t lds, const void* gaddr) {
    asm volatile("global_load_async_to_lds_b128 %0, %1, off"
                 :: "v"(lds), "v"((uint64_t)(uintptr_t)gaddr)
                 : "memory");
}

static __device__ __forceinline__ void wait_async0() {
    asm volatile("s_wait_asynccnt 0x0" ::: "memory");
}

// ---------------------------------------------------------------------------
// Pre-pass: W fp32 [K,N] (per expert, blockIdx.z) -> Wt bf16 [N,K].
// 32x32 tiles via padded LDS; coalesced on both sides.
// ---------------------------------------------------------------------------
__global__ __launch_bounds__(256) void transpose_bf16_kernel(
    const float* __restrict__ W, uint16_t* __restrict__ Wt, int K, int N)
{
    __shared__ uint16_t tile[32][33];
    const size_t eoff = (size_t)blockIdx.z * (size_t)K * (size_t)N;
    const float*  Ws = W  + eoff;
    uint16_t*     Wd = Wt + eoff;
    const int n0 = blockIdx.x * 32;
    const int k0 = blockIdx.y * 32;
    const int tx = threadIdx.x & 31;
    const int ty = threadIdx.x >> 5;
#pragma unroll
    for (int r = 0; r < 32; r += 8)
        tile[ty + r][tx] = f2bf(Ws[(size_t)(k0 + ty + r) * N + n0 + tx]);
    __syncthreads();
#pragma unroll
    for (int r = 0; r < 32; r += 8)
        Wd[(size_t)(n0 + ty + r) * K + k0 + tx] = tile[tx][ty + r];
}

// ---------------------------------------------------------------------------
// Kernel: LayerNorm + router (softmax, top-2, renorm) + aux-loss partials.
// One wave32 per token; 8 tokens per 256-thread block.
// ---------------------------------------------------------------------------
__global__ __launch_bounds__(256) void ln_gate_kernel(
    const float* __restrict__ x,       // [T, D]
    const float* __restrict__ gamma,   // [D]
    const float* __restrict__ beta,    // [D]
    const float* __restrict__ Wg,      // [D, 4]
    const float* __restrict__ bg,      // [4]
    uint16_t*    __restrict__ xn,      // [T, D] bf16 out
    float*       __restrict__ combine, // [T, 4] out
    float*       __restrict__ accum)   // [8]: counts[4], sumprob[4]
{
    const int lane  = threadIdx.x & 31;
    const int wave  = threadIdx.x >> 5;
    const int token = blockIdx.x * 8 + wave;

    const float4* x4 = (const float4*)(x + (size_t)token * DD);

    float4 xf[8];
    float sum = 0.f, sumsq = 0.f;
#pragma unroll
    for (int i = 0; i < 8; ++i) {
        float4 f = x4[lane + 32 * i];
        xf[i] = f;
        sum   += f.x + f.y + f.z + f.w;
        sumsq += f.x*f.x + f.y*f.y + f.z*f.z + f.w*f.w;
    }
#pragma unroll
    for (int off = 16; off > 0; off >>= 1) {
        sum   += __shfl_xor(sum,   off, 32);
        sumsq += __shfl_xor(sumsq, off, 32);
    }
    const float mu   = sum * (1.0f / DD);
    const float var  = sumsq * (1.0f / DD) - mu * mu;
    const float rstd = rsqrtf(var + 1e-6f);

    const float4* g4  = (const float4*)gamma;
    const float4* b4  = (const float4*)beta;
    const float4* wg4 = (const float4*)Wg;   // Wg row (4 experts) == one float4
    uint2* xn2 = (uint2*)(xn + (size_t)token * DD);

    float l0 = 0.f, l1 = 0.f, l2 = 0.f, l3 = 0.f;
#pragma unroll
    for (int i = 0; i < 8; ++i) {
        const int v4i = lane + 32 * i;       // float4 index within row
        float4 g = g4[v4i], b = b4[v4i], f = xf[i];
        float n0 = (f.x - mu) * rstd * g.x + b.x;
        float n1 = (f.y - mu) * rstd * g.y + b.y;
        float n2 = (f.z - mu) * rstd * g.z + b.z;
        float n3 = (f.w - mu) * rstd * g.w + b.w;
        uint2 packed;
        packed.x = (uint32_t)f2bf(n0) | ((uint32_t)f2bf(n1) << 16);
        packed.y = (uint32_t)f2bf(n2) | ((uint32_t)f2bf(n3) << 16);
        xn2[v4i] = packed;
        float4 w0 = wg4[v4i * 4 + 0];
        float4 w1 = wg4[v4i * 4 + 1];
        float4 w2 = wg4[v4i * 4 + 2];
        float4 w3 = wg4[v4i * 4 + 3];
        l0 += n0 * w0.x + n1 * w1.x + n2 * w2.x + n3 * w3.x;
        l1 += n0 * w0.y + n1 * w1.y + n2 * w2.y + n3 * w3.y;
        l2 += n0 * w0.z + n1 * w1.z + n2 * w2.z + n3 * w3.z;
        l3 += n0 * w0.w + n1 * w1.w + n2 * w2.w + n3 * w3.w;
    }
#pragma unroll
    for (int off = 16; off > 0; off >>= 1) {
        l0 += __shfl_xor(l0, off, 32);
        l1 += __shfl_xor(l1, off, 32);
        l2 += __shfl_xor(l2, off, 32);
        l3 += __shfl_xor(l3, off, 32);
    }

    if (lane == 0) {
        float p[4] = { l0 + bg[0], l1 + bg[1], l2 + bg[2], l3 + bg[3] };
        float mx = p[0];
#pragma unroll
        for (int e = 1; e < 4; ++e) mx = fmaxf(mx, p[e]);
        float s = 0.f;
#pragma unroll
        for (int e = 0; e < 4; ++e) { p[e] = __expf(p[e] - mx); s += p[e]; }
        float inv = 1.0f / s;
#pragma unroll
        for (int e = 0; e < 4; ++e) p[e] *= inv;

        int i1 = 0;
#pragma unroll
        for (int e = 1; e < 4; ++e) if (p[e] > p[i1]) i1 = e;
        int i2 = (i1 == 0) ? 1 : 0;
#pragma unroll
        for (int e = 0; e < 4; ++e) if (e != i1 && p[e] > p[i2]) i2 = e;

        float denom = p[i1] + p[i2];
        float c[4] = {0.f, 0.f, 0.f, 0.f};
        c[i1] = p[i1] / denom;
        c[i2] = p[i2] / denom;
        float* crow = combine + (size_t)token * 4;
#pragma unroll
        for (int e = 0; e < 4; ++e) crow[e] = c[e];

        atomicAdd(&accum[i1], 1.0f);
        atomicAdd(&accum[i2], 1.0f);
#pragma unroll
        for (int e = 0; e < 4; ++e) atomicAdd(&accum[4 + e], p[e]);
    }
}

// ---------------------------------------------------------------------------
// Tiled bf16 WMMA GEMM, async-LDS double-buffered.
// C-tile 128(M) x 128(N) per 256-thread block (8 waves); each wave computes
// a 32x64 region as 2x4 v_wmma_f32_16x16x32_bf16 accumulators.
// A: bf16 [T, K] row-major.  Bt: bf16 [N, K] row-major (pre-transposed).
// MODE 0: h = bf16(gelu(A*W1 + b1))
// MODE 1: out (+)= combine[:,e] * (A*W2 + b2)   (first expert writes)
// ---------------------------------------------------------------------------
template <int MODE>
__global__ __launch_bounds__(256) void moe_gemm_kernel(
    const uint16_t* __restrict__ A,  int lda,     // [T, K] bf16
    const uint16_t* __restrict__ Bt, int ldb,     // [N, K] bf16
    const float*    __restrict__ bias,            // [N]
    uint16_t*       __restrict__ Hout,            // MODE 0 output
    float*          __restrict__ Out,             // MODE 1 output [T, 2048]
    const float*    __restrict__ combine,         // [T, 4]
    int expert, int first, int Ksize)
{
    // double-buffered tiles: [buf][row 0..127][k 0..31], bf16
    __shared__ __align__(16) uint16_t As[2][128 * 32];
    __shared__ __align__(16) uint16_t Bs[2][128 * 32];

    const int nbase = blockIdx.x * 128;
    const int mbase = blockIdx.y * 128;
    const int tid   = threadIdx.x;
    const int lane  = tid & 31;
    const int wave  = tid >> 5;
    const int wm    = wave & 3;     // 0..3 -> M sub-tile of 32
    const int wn    = wave >> 2;    // 0..1 -> N sub-tile of 64

    const v8f vzero = {0.f, 0.f, 0.f, 0.f, 0.f, 0.f, 0.f, 0.f};
    v8f acc[2][4];
#pragma unroll
    for (int mi = 0; mi < 2; ++mi)
#pragma unroll
        for (int ni = 0; ni < 4; ++ni) acc[mi][ni] = vzero;

    const int lr = lane & 15;       // row/col within 16-wide fragment
    const int kh = lane >> 4;       // which K-half this lane holds

    // staging: each thread owns 32B (2 x b128) of A and of Bt per k-tile
    const int srow  = tid >> 1;     // 0..127
    const int shalf = tid & 1;      // 0..1
    const uint16_t* aSrc = A  + (size_t)(mbase + srow) * lda + shalf * 16;
    const uint16_t* bSrc = Bt + (size_t)(nbase + srow) * ldb + shalf * 16;
    uint32_t aLds[2], bLds[2];
    aLds[0] = lds_addr32(&As[0][srow * 32 + shalf * 16]);
    aLds[1] = lds_addr32(&As[1][srow * 32 + shalf * 16]);
    bLds[0] = lds_addr32(&Bs[0][srow * 32 + shalf * 16]);
    bLds[1] = lds_addr32(&Bs[1][srow * 32 + shalf * 16]);

    const int ntiles = Ksize >> 5;  // K-step = 32

    // prologue: async-stage tile 0 into buffer 0
    async_load_b128(aLds[0],      aSrc);
    async_load_b128(aLds[0] + 16, aSrc + 8);
    async_load_b128(bLds[0],      bSrc);
    async_load_b128(bLds[0] + 16, bSrc + 8);

    for (int t = 0; t < ntiles; ++t) {
        wait_async0();        // this wave's async copies landed in LDS
        __syncthreads();      // everyone's copies visible; prior reads done

        // async-stage tile t+1 into the other buffer (overlaps WMMA below)
        if (t + 1 < ntiles) {
            const uint16_t* aN = aSrc + (size_t)(t + 1) * 32;
            const uint16_t* bN = bSrc + (size_t)(t + 1) * 32;
            const uint32_t la = aLds[(t + 1) & 1];
            const uint32_t lb = bLds[(t + 1) & 1];
            async_load_b128(la,      aN);
            async_load_b128(la + 16, aN + 8);
            async_load_b128(lb,      bN);
            async_load_b128(lb + 16, bN + 8);
        }

        // compute current tile
        const uint4* As4 = (const uint4*)As[t & 1];
        const uint4* Bs4 = (const uint4*)Bs[t & 1];
        FragBF a[2], b[4];
#pragma unroll
        for (int mi = 0; mi < 2; ++mi) {
            const int r = wm * 32 + mi * 16 + lr;
            a[mi].q[0] = As4[r * 4 + kh];          // K = kh*8 .. +7
            a[mi].q[1] = As4[r * 4 + 2 + kh];      // K = 16+kh*8 ..
        }
#pragma unroll
        for (int ni = 0; ni < 4; ++ni) {
            const int c = wn * 64 + ni * 16 + lr;
            b[ni].q[0] = Bs4[c * 4 + kh];          // K = kh*8 .. +7
            b[ni].q[1] = Bs4[c * 4 + 2 + kh];      // K = 16+kh*8 ..
        }
#pragma unroll
        for (int mi = 0; mi < 2; ++mi)
#pragma unroll
            for (int ni = 0; ni < 4; ++ni)
                acc[mi][ni] = __builtin_amdgcn_wmma_f32_16x16x32_bf16(
                    false, a[mi].v, false, b[ni].v, (short)0, acc[mi][ni],
                    false, false);
    }

    // ---- epilogue: C layout = lanes 0-15: M=r, N=lane; lanes 16-31: M=8+r ----
    const int hi = lane >> 4;
#pragma unroll
    for (int mi = 0; mi < 2; ++mi) {
#pragma unroll
        for (int ni = 0; ni < 4; ++ni) {
            const int rowb = mbase + wm * 32 + mi * 16 + hi * 8;
            const int col  = nbase + wn * 64 + ni * 16 + lr;
            const float bv = bias[col];
#pragma unroll
            for (int r = 0; r < 8; ++r) {
                const int row = rowb + r;
                const float val = acc[mi][ni][r] + bv;
                if (MODE == 0) {
                    Hout[(size_t)row * HH + col] = f2bf(gelu_exact(val));
                } else {
                    const float w = combine[(size_t)row * 4 + expert];
                    float* o = Out + (size_t)row * HH + col;
                    *o = first ? (w * val) : (*o + w * val);
                }
            }
        }
    }
}

// ---------------------------------------------------------------------------
// Aux loss finalize: aux = E * sum_e (counts_e / (T*K)) * (sumprob_e / T)
// ---------------------------------------------------------------------------
__global__ void aux_loss_kernel(const float* __restrict__ accum,
                                float* __restrict__ out_aux)
{
    float s = 0.f;
#pragma unroll
    for (int e = 0; e < 4; ++e) {
        float frac  = accum[e]     / ((float)T_TOK * 2.0f);
        float meanp = accum[4 + e] / (float)T_TOK;
        s += frac * meanp;
    }
    *out_aux = 4.0f * s;
}

// ---------------------------------------------------------------------------
extern "C" void kernel_launch(void* const* d_in, const int* in_sizes, int n_in,
                              void* d_out, int out_size, void* d_ws, size_t ws_size,
                              hipStream_t stream)
{
    const float* x     = (const float*)d_in[0];   // [8,576,1024]
    const float* gamma = (const float*)d_in[1];   // [1024]
    const float* beta  = (const float*)d_in[2];   // [1024]
    const float* Wg    = (const float*)d_in[3];   // [1024,4]
    const float* bg    = (const float*)d_in[4];   // [4]
    const float* W1    = (const float*)d_in[5];   // [4,1024,2048]
    const float* b1    = (const float*)d_in[6];   // [4,2048]
    const float* W2    = (const float*)d_in[7];   // [4,2048,2048]
    const float* b2    = (const float*)d_in[8];   // [4,2048]
    float* out = (float*)d_out;                   // [4608*2048] results + [1] aux

    // workspace layout (all 16B aligned)
    uint16_t* p    = (uint16_t*)d_ws;
    uint16_t* xn   = p;  p += (size_t)T_TOK * DD;         //  9.4 MB
    uint16_t* hbuf = p;  p += (size_t)T_TOK * HH;         // 18.9 MB
    uint16_t* w1t  = p;  p += (size_t)NE * DD * HH;       // 16.8 MB  [E][H][D]
    uint16_t* w2t  = p;  p += (size_t)NE * HH * HH;       // 33.6 MB  [E][H][H]
    float* combine = (float*)p;                           // T*4 f32
    float* accum   = combine + (size_t)T_TOK * 4;         // 8 f32

    hipMemsetAsync(accum, 0, 8 * sizeof(float), stream);

    // one-time weight transpose+convert: fp32 [K,N] -> bf16 [N,K]
    transpose_bf16_kernel<<<dim3(HH / 32, DD / 32, NE), 256, 0, stream>>>(
        W1, w1t, DD, HH);
    transpose_bf16_kernel<<<dim3(HH / 32, HH / 32, NE), 256, 0, stream>>>(
        W2, w2t, HH, HH);

    ln_gate_kernel<<<T_TOK / 8, 256, 0, stream>>>(
        x, gamma, beta, Wg, bg, xn, combine, accum);

    dim3 grid(HH / 128, T_TOK / 128);
    for (int e = 0; e < NE; ++e) {
        moe_gemm_kernel<0><<<grid, 256, 0, stream>>>(
            xn, DD, w1t + (size_t)e * DD * HH, DD, b1 + (size_t)e * HH,
            hbuf, nullptr, nullptr, e, 0, DD);
        moe_gemm_kernel<1><<<grid, 256, 0, stream>>>(
            hbuf, HH, w2t + (size_t)e * HH * HH, HH, b2 + (size_t)e * HH,
            nullptr, out, combine, e, (e == 0) ? 1 : 0, HH);
    }

    aux_loss_kernel<<<1, 1, 0, stream>>>(accum, out + (size_t)T_TOK * HH);
}
